// Cross_Encoder_45380624450000
// MI455X (gfx1250) — compile-verified, matrix operand order checked by reading, not run on hardware
//
#include <hip/hip_runtime.h>
#include <hip/hip_bf16.h>
#include <math.h>

// ---------------- problem constants (match reference) ----------------
#define LNUM 5
#define EDIM 256
#define HNUM 16
#define DDIM 16
#define FFD  512
#define MSD  16
#define BSZ  8
#define NSEQ 200
#define BN   (BSZ * NSEQ)       // 1600 rows, divisible by 16
#define HD   (HNUM * DDIM)      // 256
#define NT   13                 // ceil(200/16)
#define NPAD (NT * 16)          // 208
#define NPAD32 224              // pad j-axis to multiple of 32 for P*V WMMA
#define SCALEF 0.25f            // 1/sqrt(16)
#define EPSV 1e-5f

typedef __attribute__((ext_vector_type(16))) __bf16 bf16x16;
typedef __attribute__((ext_vector_type(8)))  __bf16 bf16x8;
typedef __attribute__((ext_vector_type(4)))  __bf16 bf16x4;
typedef __attribute__((ext_vector_type(8)))  float  f32x8;

// ---------------- CDNA5 helpers ----------------

// Async global->LDS copy (ASYNCcnt-tracked). LDS offset = low 32 bits of the
// generic pointer (ISA 10.2: LDS aperture).
__device__ __forceinline__ void async_b128(__bf16* lds, const __bf16* g) {
  unsigned off = (unsigned)(unsigned long long)lds;
  unsigned long long ga = (unsigned long long)g;
  asm volatile("global_load_async_to_lds_b128 %0, %1, off"
               :: "v"(off), "v"(ga) : "memory");
}
__device__ __forceinline__ void wait_async0() {
  asm volatile("s_wait_asynccnt 0" ::: "memory");
}
__device__ __forceinline__ void wait_ds0() {
  asm volatile("s_wait_dscnt 0" ::: "memory");
}

__device__ __forceinline__ bf16x8 bz8() {
  bf16x8 z;
#pragma unroll
  for (int i = 0; i < 8; ++i) z[i] = (__bf16)0.f;
  return z;
}
// ISA 7.12.2 16-bit fragment layout: each half of a lane's 16 elements is a
// run of 8 consecutive K values -> fragment == two 16B contiguous loads.
//   elements 0..7 : K = half*8 + 0..7
//   elements 8..15: K = 16 + half*8 + 0..7      (half = lane>>4)
__device__ __forceinline__ bf16x16 mkfrag(bf16x8 lo, bf16x8 hi) {
  return __builtin_shufflevector(lo, hi, 0, 1, 2, 3, 4, 5, 6, 7,
                                 8, 9, 10, 11, 12, 13, 14, 15);
}

// ---------------- preprocessing kernels ----------------

// Per-batch min/range; writes zero-padded scaled (B,208,208) in BOTH
// orientations so the attention kernel has no bounds checks and no
// transpose select.
__global__ __launch_bounds__(256) void k_scale(const float* __restrict__ data,
                                               float* __restrict__ sp,
                                               float* __restrict__ spt) {
  __shared__ float smn[256], smx[256];
  int b = blockIdx.x, t = threadIdx.x;
  const float* p = data + (size_t)b * NSEQ * NSEQ;
  float mn = 3.4e38f, mx = -3.4e38f;
  for (int i = t; i < NSEQ * NSEQ; i += 256) {
    float v = p[i]; mn = fminf(mn, v); mx = fmaxf(mx, v);
  }
  smn[t] = mn; smx[t] = mx;
  __syncthreads();
  for (int s = 128; s > 0; s >>= 1) {
    if (t < s) { smn[t] = fminf(smn[t], smn[t + s]); smx[t] = fmaxf(smx[t], smx[t + s]); }
    __syncthreads();
  }
  float bmn = smn[0];
  float rng = smx[0] - bmn;
  if (rng == 0.f) rng = 1.f;
  float inv = 1.f / rng;
  float* op  = sp  + (size_t)b * NPAD * NPAD;
  float* opt = spt + (size_t)b * NPAD * NPAD;
  for (int idx = t; idx < NPAD * NPAD; idx += 256) {
    int i = idx / NPAD, j = idx % NPAD;
    float v = (i < NSEQ && j < NSEQ) ? (p[i * NSEQ + j] - bmn) * inv : 0.f;
    op[idx] = v;
    opt[j * NPAD + i] = v;
  }
}

__global__ __launch_bounds__(256) void k_row_init(const float* __restrict__ nr,
                                                  const float* __restrict__ nW,
                                                  const float* __restrict__ nB,
                                                  float* rf, __bf16* rb,
                                                  float* cf, __bf16* cb) {
  int idx = blockIdx.x * 256 + threadIdx.x;
  if (idx >= BN * EDIM) return;
  int e = idx & (EDIM - 1);
  int bi = idx >> 8;
  float v = nr[bi] * nW[e] + nB[e];
  __bf16 h = (__bf16)v;
  rf[idx] = v; cf[idx] = v; rb[idx] = h; cb[idx] = h;
}

// weight convert + transpose: src (L*2, K, Nc) fp32 -> dst (L*2, Nc, K) bf16
__global__ __launch_bounds__(256) void k_cvtT(const float* __restrict__ s,
                                              __bf16* __restrict__ d,
                                              int K, int Nc) {
  int idx = blockIdx.x * 256 + threadIdx.x;
  int per = K * Nc;
  if (idx >= LNUM * 2 * per) return;
  int ls = idx / per, r = idx % per;
  int n = r / K, k = r % K;
  d[idx] = (__bf16)s[(size_t)ls * per + (size_t)k * Nc + n];
}

// V transpose for P@V B-fragments: vT[b,h,d,j] (j padded to 224 with zeros)
__global__ __launch_bounds__(256) void k_vt(const __bf16* __restrict__ v,
                                            __bf16* __restrict__ vt) {
  int idx = blockIdx.x * 256 + threadIdx.x;
  if (idx >= BSZ * HNUM * DDIM * NPAD32) return;
  int j = idx % NPAD32;
  int rest = idx / NPAD32;
  int d = rest % DDIM;
  int bh = rest / DDIM;
  int h = bh % HNUM, b = bh / HNUM;
  vt[idx] = (j < NSEQ) ? v[((size_t)(b * NSEQ + j)) * HD + h * DDIM + d]
                       : (__bf16)0.f;
}

// ---------------- WMMA GEMM: C[16x64 per wave] = act(A@W + bias + res) ----
// A: (M,K) bf16 row-major; WT: (Nc,K) bf16 (pre-transposed weights).
// LDS-staged via async b128 loads; fragments read as 2x ds_load_b128.
template <bool HAS_BIAS, bool HAS_RES, bool RELU, bool OUTF, bool OUTB>
__global__ __launch_bounds__(32) void k_gemm(const __bf16* __restrict__ A, int lda,
                                             const __bf16* __restrict__ WT,
                                             const float* __restrict__ bias,
                                             const float* __restrict__ res,
                                             float* __restrict__ outf,
                                             __bf16* __restrict__ outb,
                                             int K, int ldo) {
  __shared__ __bf16 As[16 * 32];   // 16 rows x 32 k
  __shared__ __bf16 Bs[64 * 32];   // 64 n   x 32 k (n-major)
  const int lane = threadIdx.x & 31;
  const int m0 = blockIdx.x << 4;
  const int n0 = blockIdx.y << 6;
  const int half8 = (lane & 16) ? 8 : 0;
  const int mloc = lane & 15;
  f32x8 acc[4];
#pragma unroll
  for (int t = 0; t < 4; ++t)
    acc[t] = (f32x8){0.f, 0.f, 0.f, 0.f, 0.f, 0.f, 0.f, 0.f};

  for (int k0 = 0; k0 < K; k0 += 32) {
    wait_ds0();  // prior LDS fragment reads done before async overwrite
    {  // A tile: 64 x 16B chunks, 2 per lane
      int id = lane << 1;
      int r = id >> 2, c8 = (id & 3) << 3;
      async_b128(&As[r * 32 + c8], A + (size_t)(m0 + r) * lda + k0 + c8);
      int r2 = (id + 1) >> 2, c82 = ((id + 1) & 3) << 3;
      async_b128(&As[r2 * 32 + c82], A + (size_t)(m0 + r2) * lda + k0 + c82);
    }
#pragma unroll
    for (int c = 0; c < 8; ++c) {  // B tile: 256 x 16B chunks, 8 per lane
      int id = lane * 8 + c;
      int nl = id >> 2, c8 = (id & 3) << 3;
      async_b128(&Bs[nl * 32 + c8], WT + (size_t)(n0 + nl) * K + k0 + c8);
    }
    wait_async0();
    __syncthreads();
    bf16x16 a = mkfrag(*(const bf16x8*)(As + mloc * 32 + half8),
                       *(const bf16x8*)(As + mloc * 32 + 16 + half8));
#pragma unroll
    for (int t = 0; t < 4; ++t) {
      const __bf16* bp = Bs + (t * 16 + mloc) * 32;
      bf16x16 b = mkfrag(*(const bf16x8*)(bp + half8),
                         *(const bf16x8*)(bp + 16 + half8));
      acc[t] = __builtin_amdgcn_wmma_f32_16x16x32_bf16(false, a, false, b,
                                                       (short)0, acc[t],
                                                       false, false);
    }
    __syncthreads();
  }

  const int m_out = m0 + ((lane < 16) ? 0 : 8);
#pragma unroll
  for (int t = 0; t < 4; ++t) {
    int n = n0 + t * 16 + mloc;
    float bv = HAS_BIAS ? bias[n] : 0.f;
#pragma unroll
    for (int r = 0; r < 8; ++r) {
      int m = m_out + r;
      size_t o = (size_t)m * ldo + n;
      float v = acc[t][r] + bv;
      if (HAS_RES) v += res[o];
      if (RELU) v = v > 0.f ? v : 0.f;
      if (OUTF) outf[o] = v;
      if (OUTB) outb[o] = (__bf16)v;
    }
  }
}

// ---------------- edge rank-1 scalars: qw = q.edgeW_h, qb = q.edgeB_h ----
// Padded to 208 rows per (b,h) so the attention kernel loads unconditionally.
__global__ __launch_bounds__(256) void k_qscal(const float* __restrict__ qf,
                                               const float* __restrict__ eW,
                                               const float* __restrict__ eB,
                                               float* __restrict__ qw,
                                               float* __restrict__ qb) {
  int idx = blockIdx.x * 256 + threadIdx.x;  // (b*H + h)*NPAD + i
  if (idx >= BSZ * HNUM * NPAD) return;
  int i = idx % NPAD;
  int bh = idx / NPAD;
  int h = bh % HNUM, b = bh / HNUM;
  float sw = 0.f, sb = 0.f;
  if (i < NSEQ) {
    const float* q = qf + ((size_t)(b * NSEQ + i)) * HD + h * DDIM;
#pragma unroll
    for (int d = 0; d < DDIM; ++d) {
      sw += q[d] * eW[h * DDIM + d];
      sb += q[d] * eB[h * DDIM + d];
    }
  }
  qw[idx] = sw; qb[idx] = sb;
}

// ---------------- fused attention: QK^T -> MLP score -> softmax -> P@V ----
// scaled is the zero-padded (B,208,208) matrix in the orientation this block
// needs (host passes normal or transposed copy). q/k buffers over-allocated
// by 16 rows so fragment loads in the last tile are unconditional; garbage
// rows only feed scores masked to -1e30 (j>=N) or rows never stored (i>=N).
__global__ __launch_bounds__(32) void k_attn(
    const __bf16* __restrict__ qb16, const __bf16* __restrict__ kb16,
    const __bf16* __restrict__ vT, const float* __restrict__ scaled,
    const float* __restrict__ qw, const float* __restrict__ qbv,
    const float* __restrict__ m1w, const float* __restrict__ m1b,
    const float* __restrict__ m2w, const float* __restrict__ m2b,
    __bf16* __restrict__ ao) {
  __shared__ float  sc[16 * NPAD];
  __shared__ __bf16 pb[16 * NPAD32];
  const int lane = threadIdx.x & 31;
  const int bh = blockIdx.x;
  const int b = bh / HNUM, h = bh % HNUM;
  const int i0 = blockIdx.y << 4;
  const int half8 = (lane & 16) ? 8 : 0;
  const int mloc = lane & 15;

  // per-head mixed-score MLP params (wave-uniform -> scalar loads)
  float w0[MSD], w1[MSD], bbv[MSD], w2[MSD];
#pragma unroll
  for (int m = 0; m < MSD; ++m) {
    w0[m] = m1w[(h * 2 + 0) * MSD + m];
    w1[m] = m1w[(h * 2 + 1) * MSD + m];
    bbv[m] = m1b[h * MSD + m];
    w2[m] = m2w[h * MSD + m];
  }
  const float c2 = m2b[h];

  // Q fragment: D=16 -> one 16B load, upper half (K>=16) zero
  bf16x16 aq = mkfrag(
      *(const bf16x8*)(qb16 + ((size_t)(b * NSEQ + i0 + mloc)) * HD + h * DDIM + half8),
      bz8());
  // per-row rank-1 edge scalars (padded buffers -> unconditional)
  float rqw[8], rqb[8];
#pragma unroll
  for (int r = 0; r < 8; ++r) {
    int i = i0 + ((lane < 16) ? r : r + 8);
    rqw[r] = qw[(size_t)bh * NPAD + i];
    rqb[r] = qbv[(size_t)bh * NPAD + i];
  }
  const float* srow = scaled + (size_t)b * NPAD * NPAD;

  for (int jt = 0; jt < NT; ++jt) {
    int j0 = jt << 4;
    bf16x16 bk = mkfrag(
        *(const bf16x8*)(kb16 + ((size_t)(b * NSEQ + j0 + mloc)) * HD + h * DDIM + half8),
        bz8());
    f32x8 dacc = {0.f, 0.f, 0.f, 0.f, 0.f, 0.f, 0.f, 0.f};
    dacc = __builtin_amdgcn_wmma_f32_16x16x32_bf16(false, aq, false, bk,
                                                   (short)0, dacc, false, false);
    int j = j0 + mloc;
#pragma unroll
    for (int r = 0; r < 8; ++r) {
      int rl = (lane < 16) ? r : r + 8;
      int i = i0 + rl;
      float dot = dacc[r] * SCALEF;
      float es = (srow[(size_t)i * NPAD + j] * rqw[r] + rqb[r]) * SCALEF;
      float s2 = c2;
#pragma unroll
      for (int m = 0; m < MSD; ++m) {
        float t = dot * w0[m] + es * w1[m] + bbv[m];
        t = t > 0.f ? t : 0.f;
        s2 += t * w2[m];
      }
      sc[rl * NPAD + j] = (j < NSEQ) ? s2 : -1e30f;
    }
  }
  __syncthreads();

  // row softmax over j (vectorized), bf16 probabilities padded to 224
  if (lane < 16) {
    float4* scv = (float4*)(sc + lane * NPAD);
    float mx = -3.4e38f;
    for (int jv = 0; jv < NPAD / 4; ++jv) {
      float4 v = scv[jv];
      mx = fmaxf(mx, fmaxf(fmaxf(v.x, v.y), fmaxf(v.z, v.w)));
    }
    float ssum = 0.f;
    for (int jv = 0; jv < NPAD / 4; ++jv) {
      float4 v = scv[jv];
      v.x = __expf(v.x - mx); v.y = __expf(v.y - mx);
      v.z = __expf(v.z - mx); v.w = __expf(v.w - mx);
      ssum += (v.x + v.y) + (v.z + v.w);
      scv[jv] = v;
    }
    float inv = 1.f / ssum;
    for (int jv = 0; jv < NPAD / 4; ++jv) {
      float4 v = scv[jv];
      bf16x4 o;
      o[0] = (__bf16)(v.x * inv); o[1] = (__bf16)(v.y * inv);
      o[2] = (__bf16)(v.z * inv); o[3] = (__bf16)(v.w * inv);
      *(bf16x4*)(pb + lane * NPAD32 + jv * 4) = o;
    }
    bf16x4 z4;
    z4[0] = z4[1] = z4[2] = z4[3] = (__bf16)0.f;
    for (int j = NPAD; j < NPAD32; j += 4)
      *(bf16x4*)(pb + lane * NPAD32 + j) = z4;
  }
  __syncthreads();

  // out = P @ V  (K = j axis, 7 steps of 32; vT gives contiguous B-fragments)
  f32x8 oacc = {0.f, 0.f, 0.f, 0.f, 0.f, 0.f, 0.f, 0.f};
  const __bf16* vrow = vT + ((size_t)bh * DDIM + mloc) * NPAD32;
#pragma unroll
  for (int st = 0; st < NPAD32 / 32; ++st) {
    int j0 = st << 5;
    bf16x16 ap = mkfrag(*(const bf16x8*)(pb + mloc * NPAD32 + j0 + half8),
                        *(const bf16x8*)(pb + mloc * NPAD32 + j0 + 16 + half8));
    bf16x16 bv = mkfrag(*(const bf16x8*)(vrow + j0 + half8),
                        *(const bf16x8*)(vrow + j0 + 16 + half8));
    oacc = __builtin_amdgcn_wmma_f32_16x16x32_bf16(false, ap, false, bv,
                                                   (short)0, oacc, false, false);
  }
#pragma unroll
  for (int r = 0; r < 8; ++r) {
    int i = i0 + ((lane < 16) ? r : r + 8);
    if (i < NSEQ)
      ao[((size_t)(b * NSEQ + i)) * HD + h * DDIM + mloc] = (__bf16)oacc[r];
  }
}

// ---------------- InstanceNorm over nodes (axis=1), affine ----------------
__global__ __launch_bounds__(EDIM) void k_inorm(const float* __restrict__ x,
                                                const float* __restrict__ g,
                                                const float* __restrict__ be,
                                                float* __restrict__ of,
                                                __bf16* __restrict__ ob) {
  int b = blockIdx.x, e = threadIdx.x;
  float s = 0.f, s2 = 0.f;
  for (int i = 0; i < NSEQ; ++i) {
    float v = x[((size_t)(b * NSEQ + i)) * EDIM + e];
    s += v; s2 += v * v;
  }
  float mu = s * (1.f / NSEQ);
  float var = s2 * (1.f / NSEQ) - mu * mu;
  float inv = rsqrtf(var + EPSV);
  float gg = g[e], bbv = be[e];
  for (int i = 0; i < NSEQ; ++i) {
    size_t o = ((size_t)(b * NSEQ + i)) * EDIM + e;
    float v = (x[o] - mu) * inv * gg + bbv;
    of[o] = v;
    ob[o] = (__bf16)v;
  }
}

// ---------------- final output concat (row, col) ----------------
__global__ __launch_bounds__(256) void k_out(const float* __restrict__ a,
                                             const float* __restrict__ b,
                                             float* __restrict__ o) {
  int i = blockIdx.x * 256 + threadIdx.x;
  if (i < BN * EDIM) o[i] = a[i];
  else if (i < 2 * BN * EDIM) o[i] = b[i - BN * EDIM];
}

// ---------------- host orchestration ----------------
extern "C" void kernel_launch(void* const* d_in, const int* in_sizes, int n_in,
                              void* d_out, int out_size, void* d_ws, size_t ws_size,
                              hipStream_t stream) {
  (void)in_sizes; (void)n_in; (void)out_size; (void)ws_size;
  const float* data  = (const float*)d_in[0];
  const float* nrand = (const float*)d_in[1];
  const float* Wq  = (const float*)d_in[2];
  const float* Wk  = (const float*)d_in[3];
  const float* Wv  = (const float*)d_in[4];
  const float* Wc  = (const float*)d_in[5];
  const float* bc  = (const float*)d_in[6];
  const float* g1  = (const float*)d_in[7];
  const float* b1  = (const float*)d_in[8];
  const float* W1  = (const float*)d_in[9];
  const float* bW1 = (const float*)d_in[10];
  const float* W2  = (const float*)d_in[11];
  const float* bW2 = (const float*)d_in[12];
  const float* g2  = (const float*)d_in[13];
  const float* b2  = (const float*)d_in[14];
  const float* m1w = (const float*)d_in[15];
  const float* m1b = (const float*)d_in[16];
  const float* m2w = (const float*)d_in[17];
  const float* m2b = (const float*)d_in[18];
  const float* nodeW = (const float*)d_in[19];
  const float* nodeB = (const float*)d_in[20];
  const float* edgeW = (const float*)d_in[21];
  const float* edgeB = (const float*)d_in[22];

  char* ws = (char*)d_ws;
  size_t off = 0;
  auto alloc = [&](size_t bytes) -> void* {
    void* p = ws + off;
    off = (off + bytes + 255) & ~(size_t)255;
    return p;
  };

  float* scaledP  = (float*)alloc((size_t)BSZ * NPAD * NPAD * 4);
  float* scaledPT = (float*)alloc((size_t)BSZ * NPAD * NPAD * 4);
  float*  rowf[2]; __bf16* rowb[2];
  float*  colf[2]; __bf16* colb[2];
  for (int i = 0; i < 2; ++i) {
    rowf[i] = (float*)alloc((size_t)BN * EDIM * 4);
    colf[i] = (float*)alloc((size_t)BN * EDIM * 4);
    rowb[i] = (__bf16*)alloc((size_t)BN * EDIM * 2);
    colb[i] = (__bf16*)alloc((size_t)BN * EDIM * 2);
  }
  float*  qf    = (float*)alloc((size_t)BN * HD * 4);
  __bf16* qb16b = (__bf16*)alloc((size_t)(BN + 16) * HD * 2);  // +16 pad rows
  __bf16* kb16b = (__bf16*)alloc((size_t)(BN + 16) * HD * 2);  // +16 pad rows
  __bf16* vb16b = (__bf16*)alloc((size_t)BN * HD * 2);
  __bf16* vtb   = (__bf16*)alloc((size_t)BSZ * HNUM * DDIM * NPAD32 * 2);
  float*  qwbuf = (float*)alloc((size_t)BSZ * HNUM * NPAD * 4);
  float*  qbbuf = (float*)alloc((size_t)BSZ * HNUM * NPAD * 4);
  __bf16* aob   = (__bf16*)alloc((size_t)BN * HD * 2);
  float*  xbuf  = (float*)alloc((size_t)BN * EDIM * 4);
  float*  o1f   = (float*)alloc((size_t)BN * EDIM * 4);
  __bf16* o1b   = (__bf16*)alloc((size_t)BN * EDIM * 2);
  __bf16* ff1b  = (__bf16*)alloc((size_t)BN * FFD * 2);

  const size_t nWqkv = (size_t)LNUM * 2 * EDIM * HD;   // 655360
  const size_t nWff  = (size_t)LNUM * 2 * EDIM * FFD;  // 1310720
  __bf16* WqT = (__bf16*)alloc(nWqkv * 2);
  __bf16* WkT = (__bf16*)alloc(nWqkv * 2);
  __bf16* WvT = (__bf16*)alloc(nWqkv * 2);
  __bf16* WcT = (__bf16*)alloc(nWqkv * 2);
  __bf16* W1T = (__bf16*)alloc(nWff * 2);
  __bf16* W2T = (__bf16*)alloc(nWff * 2);

  // preprocessing
  k_scale<<<BSZ, 256, 0, stream>>>(data, scaledP, scaledPT);
  k_row_init<<<(BN * EDIM + 255) / 256, 256, 0, stream>>>(
      nrand, nodeW, nodeB, rowf[0], rowb[0], colf[0], colb[0]);
  int gq = (int)((nWqkv + 255) / 256), gf = (int)((nWff + 255) / 256);
  k_cvtT<<<gq, 256, 0, stream>>>(Wq, WqT, EDIM, HD);
  k_cvtT<<<gq, 256, 0, stream>>>(Wk, WkT, EDIM, HD);
  k_cvtT<<<gq, 256, 0, stream>>>(Wv, WvT, EDIM, HD);
  k_cvtT<<<gq, 256, 0, stream>>>(Wc, WcT, HD, EDIM);
  k_cvtT<<<gf, 256, 0, stream>>>(W1, W1T, EDIM, FFD);
  k_cvtT<<<gf, 256, 0, stream>>>(W2, W2T, FFD, EDIM);

  const dim3 g16(BN / 16, EDIM / 64);   // 100 x 4
  const dim3 gff(BN / 16, FFD / 64);    // 100 x 8
  const dim3 gattn(BSZ * HNUM, NT);     // 128 x 13

  int cur = 0;
  for (int l = 0; l < LNUM; ++l) {
    int nxt = cur ^ 1;
    for (int s = 0; s < 2; ++s) {
      const float*  rf_in = (s == 0) ? rowf[cur] : colf[cur];
      const __bf16* rb_in = (s == 0) ? rowb[cur] : colb[cur];
      const __bf16* cb_in = (s == 0) ? colb[cur] : rowb[cur];
      float*  rf_out = (s == 0) ? rowf[nxt] : colf[nxt];
      __bf16* rb_out = (s == 0) ? rowb[nxt] : colb[nxt];
      const float* sc_in = (s == 0) ? scaledP : scaledPT;
      size_t wo = (size_t)l * 2 + s;

      // Q/K/V projections (bf16 WMMA, K=256)
      k_gemm<false, false, false, true, true><<<g16, 32, 0, stream>>>(
          rb_in, EDIM, WqT + wo * EDIM * HD, nullptr, nullptr, qf, qb16b,
          EDIM, HD);
      k_gemm<false, false, false, false, true><<<g16, 32, 0, stream>>>(
          cb_in, EDIM, WkT + wo * EDIM * HD, nullptr, nullptr, nullptr, kb16b,
          EDIM, HD);
      k_gemm<false, false, false, false, true><<<g16, 32, 0, stream>>>(
          cb_in, EDIM, WvT + wo * EDIM * HD, nullptr, nullptr, nullptr, vb16b,
          EDIM, HD);
      // V transpose for contiguous P@V fragments
      k_vt<<<(BSZ * HNUM * DDIM * NPAD32 + 255) / 256, 256, 0, stream>>>(
          vb16b, vtb);
      // rank-1 edge-score scalars (padded to 208 rows)
      k_qscal<<<(BSZ * HNUM * NPAD + 255) / 256, 256, 0, stream>>>(
          qf, edgeW, edgeB, qwbuf, qbbuf);
      // fused attention
      k_attn<<<gattn, 32, 0, stream>>>(
          qb16b, kb16b, vtb, sc_in, qwbuf, qbbuf,
          m1w + wo * HNUM * 2 * MSD, m1b + wo * HNUM * MSD,
          m2w + wo * HNUM * MSD, m2b + wo * HNUM, aob);
      // output projection + residual
      k_gemm<true, true, false, true, false><<<g16, 32, 0, stream>>>(
          aob, HD, WcT + wo * HD * EDIM, bc + wo * EDIM, rf_in, xbuf, nullptr,
          HD, EDIM);
      k_inorm<<<BSZ, EDIM, 0, stream>>>(xbuf, g1 + wo * EDIM, b1 + wo * EDIM,
                                        o1f, o1b);
      // FFN
      k_gemm<true, false, true, false, true><<<gff, 32, 0, stream>>>(
          o1b, EDIM, W1T + wo * EDIM * FFD, bW1 + wo * FFD, nullptr, nullptr,
          ff1b, EDIM, FFD);
      k_gemm<true, true, false, true, false><<<g16, 32, 0, stream>>>(
          ff1b, FFD, W2T + wo * FFD * EDIM, bW2 + wo * EDIM, o1f, xbuf,
          nullptr, FFD, EDIM);
      k_inorm<<<BSZ, EDIM, 0, stream>>>(xbuf, g2 + wo * EDIM, b2 + wo * EDIM,
                                        rf_out, rb_out);
    }
    cur = nxt;
  }
  k_out<<<(2 * BN * EDIM + 255) / 256, 256, 0, stream>>>(rowf[cur], colf[cur],
                                                         (float*)d_out);
}
